// RGCN_335007449370
// MI455X (gfx1250) — compile-verified, matrix operand order checked by reading, not run on hardware
//
#include <hip/hip_runtime.h>

// ---------------------------------------------------------------------------
// RGCN (2-layer, mean aggregation, R=8 relations) for MI455X / gfx1250.
//
//   per layer:
//     out = in @ root + b                         (WMMA strip pass, write)
//     for r in 0..7:
//       msum = 0;  msum[dst] += in[src]  (rel r)  (f32 atomics, L2-resident)
//       out += (msum/max(cnt,1)) @ W[r]           (WMMA strip pass, accumulate)
//     relu (layer 1)
//
// Weights are pre-converted once per call into bf16 laid out EXACTLY as the
// v_wmma_f32_16x16x32_bf16 B operand wants per lane, so the inner loop's B
// load is a single contiguous 32B load (2x global_load_b128), no address ALU.
// One wave computes a 16-row x full-width strip: A fragment built once per
// k-step and reused across all column tiles (8 for D=128, 4 for D=64).
// ---------------------------------------------------------------------------

typedef __attribute__((ext_vector_type(16))) __bf16 v16bf;
typedef __attribute__((ext_vector_type(8)))  float  v8f;

__global__ void zero_f32(float* __restrict__ p, long n) {
    long i = (long)blockIdx.x * blockDim.x + threadIdx.x;
    if (i < n) p[i] = 0.0f;
}

__global__ void relu_f32(float* __restrict__ p, long n) {
    long i = (long)blockIdx.x * blockDim.x + threadIdx.x;
    if (i < n) p[i] = fmaxf(p[i], 0.0f);
}

__global__ void edge_count(const int* __restrict__ dst, const int* __restrict__ et,
                           int E, float* __restrict__ cnt) {
    int e = blockIdx.x * blockDim.x + threadIdx.x;
    if (e < E) atomicAdd(&cnt[(long)dst[e] * 8 + et[e]], 1.0f);
}

// One wave per edge; each lane scatters 4 consecutive floats.
__global__ void edge_aggregate(const float* __restrict__ X,
                               const int* __restrict__ src,
                               const int* __restrict__ dst,
                               const int* __restrict__ et,
                               int E, int rel,
                               float* __restrict__ msum) {
    long t = (long)blockIdx.x * blockDim.x + threadIdx.x;
    long e = t >> 5;
    int lane = (int)(t & 31);
    if (e >= E) return;
    if (et[e] != rel) return;
    const float4 v = *(const float4*)(X + (long)src[e] * 128 + lane * 4);
    float* o = msum + (long)dst[e] * 128 + lane * 4;
    atomicAdd(o + 0, v.x);
    atomicAdd(o + 1, v.y);
    atomicAdd(o + 2, v.z);
    atomicAdd(o + 3, v.w);
}

// Convert W (K=128 x Ncols f32, row-major) into bf16 in per-lane WMMA B
// layout:  Bsw[((ct*4 + kt)*32 + lane)*16 + i]  where vgpr pair v = i/2,
// half = lane>>4, k = kt*32 + half*8 + (v<4 ? 2v : 8+2v) + (i&1),
// col = ct*16 + (lane&15).
__global__ void swizzle_B(const float* __restrict__ W, int ldw, int nct,
                          __bf16* __restrict__ Bsw) {
    int t = blockIdx.x * blockDim.x + threadIdx.x;
    if (t >= nct * 4 * 32 * 16) return;
    int i    = t & 15;
    int lane = (t >> 4) & 31;
    int kt   = (t >> 9) & 3;
    int ct   = t >> 11;
    int half = lane >> 4, ln = lane & 15;
    int v = i >> 1, h = i & 1;
    int k = kt * 32 + half * 8 + (v < 4 ? 2 * v : 8 + 2 * v) + h;
    Bsw[t] = (__bf16)W[(long)k * ldw + ct * 16 + ln];
}

// One wave32 per 16-row strip, NCT 16-col tiles wide.  K = 128.
// gfx1250 wave32 WMMA layouts (ISA 7.12.2):
//   A per-lane K pairs: vgpr v holds k = (v<4 ? 2v : 8+2v) + 8*(lane>=16) (+pair)
//   C/D: vgpr j -> row j + 8*(lane>=16), col = lane%16
template <int NCT>
__launch_bounds__(32)
__global__ void rgcn_gemm_strip(const float* __restrict__ Asrc, int lda,
                                const __bf16* __restrict__ Bsw,
                                const float* __restrict__ cnt, int rel,
                                const float* __restrict__ bias,
                                float* __restrict__ out, int ldo,
                                int Nn, int accumulate) {
    const int lane = threadIdx.x;
    const int half = lane >> 4;
    const int ln   = lane & 15;
    const int row0 = blockIdx.x * 16;

    int node = row0 + ln;
    if (node >= Nn) node = Nn - 1;            // clamp loads; stores are masked
    float scale = 1.0f;
    if (cnt) scale = 1.0f / fmaxf(cnt[(long)node * 8 + rel], 1.0f);

    v8f c[NCT];
    if (accumulate) {
#pragma unroll
        for (int ct = 0; ct < NCT; ++ct)
#pragma unroll
            for (int j = 0; j < 8; ++j) {
                int m = row0 + j + half * 8;
                c[ct][j] = (m < Nn) ? out[(long)m * ldo + ct * 16 + ln] : 0.0f;
            }
    } else {
#pragma unroll
        for (int ct = 0; ct < NCT; ++ct) {
            float bv = bias ? bias[ct * 16 + ln] : 0.0f;
#pragma unroll
            for (int j = 0; j < 8; ++j) c[ct][j] = bv;
        }
    }

    const float* arow = Asrc + (long)node * lda;
#pragma unroll
    for (int kt = 0; kt < 4; ++kt) {
        v16bf a;
        const int k0 = kt * 32 + half * 8;
#pragma unroll
        for (int v = 0; v < 8; ++v) {
            const int kb = k0 + (v < 4 ? 2 * v : 8 + 2 * v);
            const float2 af = *(const float2*)(arow + kb);
            a[2 * v]     = (__bf16)(af.x * scale);
            a[2 * v + 1] = (__bf16)(af.y * scale);
        }
#pragma unroll
        for (int ct = 0; ct < NCT; ++ct) {
            const v16bf b =
                *(const v16bf*)(Bsw + ((size_t)(ct * 4 + kt) * 32 + lane) * 16);
            c[ct] = __builtin_amdgcn_wmma_f32_16x16x32_bf16(
                        /*neg_a=*/false, a, /*neg_b=*/false, b,
                        /*c_mod=*/(short)0, c[ct],
                        /*reuse_a=*/false, /*reuse_b=*/false);
        }
    }

#pragma unroll
    for (int ct = 0; ct < NCT; ++ct)
#pragma unroll
        for (int j = 0; j < 8; ++j) {
            int m = row0 + j + half * 8;
            if (m < Nn) out[(long)m * ldo + ct * 16 + ln] = c[ct][j];
        }
}

extern "C" void kernel_launch(void* const* d_in, const int* in_sizes, int n_in,
                              void* d_out, int out_size, void* d_ws, size_t ws_size,
                              hipStream_t stream) {
    (void)n_in; (void)out_size; (void)ws_size;

    const float* x     = (const float*)d_in[0];   // [N,128]
    const float* W1    = (const float*)d_in[1];   // [8,128,128]
    const float* root1 = (const float*)d_in[2];   // [128,128]
    const float* b1    = (const float*)d_in[3];   // [128]
    const float* W2    = (const float*)d_in[4];   // [8,128,64]
    const float* root2 = (const float*)d_in[5];   // [128,64]
    const float* b2    = (const float*)d_in[6];   // [64]
    const int*   ei    = (const int*)d_in[7];     // [2,E]
    const int*   et    = (const int*)d_in[8];     // [E]

    const int DIN = 128, DHID = 128, DOUT = 64, R = 8;
    const int Nn = in_sizes[0] / DIN;
    const int E  = in_sizes[7] / 2;
    const int* src = ei;
    const int* dst = ei + E;

    // ws layout: cnt[N*8]f32 | msum[N*128]f32 | h[N*128]f32 |
    //            Bsw1[9*16384]bf16 | Bsw2[9*8192]bf16          (~53.3 MB)
    float*  cnt  = (float*)d_ws;
    float*  msum = cnt  + (size_t)Nn * R;
    float*  h    = msum + (size_t)Nn * DIN;
    __bf16* bsw1 = (__bf16*)(h + (size_t)Nn * DHID);
    __bf16* bsw2 = bsw1 + (size_t)9 * DIN * DHID;   // 9 matrices of 128x128
    float*  outp = (float*)d_out;

    const int rowT = (Nn + 15) / 16;
    const long aggT = (long)E * 32;
    const int SW1 = DIN * DHID;   // 16384 bf16 per swizzled 128x128 matrix
    const int SW2 = DHID * DOUT;  // 8192 per 128x64

    // ---- pre-swizzle weights to bf16 WMMA B layout (tiny, once per call) ----
    for (int r = 0; r < R; ++r)
        swizzle_B<<<SW1 / 256, 256, 0, stream>>>(
            W1 + (size_t)r * SW1, DHID, DHID / 16, bsw1 + (size_t)r * SW1);
    swizzle_B<<<SW1 / 256, 256, 0, stream>>>(root1, DHID, DHID / 16,
                                             bsw1 + (size_t)8 * SW1);
    for (int r = 0; r < R; ++r)
        swizzle_B<<<SW2 / 256, 256, 0, stream>>>(
            W2 + (size_t)r * SW2, DOUT, DOUT / 16, bsw2 + (size_t)r * SW2);
    swizzle_B<<<SW2 / 256, 256, 0, stream>>>(root2, DOUT, DOUT / 16,
                                             bsw2 + (size_t)8 * SW2);

    // ---- shared edge counts (same graph for both layers) ----
    zero_f32<<<(unsigned)(((long)Nn * R + 255) / 256), 256, 0, stream>>>(
        cnt, (long)Nn * R);
    edge_count<<<(unsigned)((E + 255) / 256), 256, 0, stream>>>(dst, et, E, cnt);

    // ---- layer 1: h = relu( sum_r mean_r @ W1[r] + x @ root1 + b1 ) ----
    rgcn_gemm_strip<8><<<rowT, 32, 0, stream>>>(
        x, DIN, bsw1 + (size_t)8 * SW1, nullptr, 0, b1, h, DHID, Nn, 0);
    for (int r = 0; r < R; ++r) {
        zero_f32<<<(unsigned)(((long)Nn * DIN + 255) / 256), 256, 0, stream>>>(
            msum, (long)Nn * DIN);
        edge_aggregate<<<(unsigned)((aggT + 255) / 256), 256, 0, stream>>>(
            x, src, dst, et, E, r, msum);
        rgcn_gemm_strip<8><<<rowT, 32, 0, stream>>>(
            msum, DIN, bsw1 + (size_t)r * SW1, cnt, r, nullptr, h, DHID, Nn, 1);
    }
    relu_f32<<<(unsigned)(((long)Nn * DHID + 255) / 256), 256, 0, stream>>>(
        h, (long)Nn * DHID);

    // ---- layer 2: out = sum_r mean_r @ W2[r] + h @ root2 + b2 ----
    rgcn_gemm_strip<4><<<rowT, 32, 0, stream>>>(
        h, DHID, bsw2 + (size_t)8 * SW2, nullptr, 0, b2, outp, DOUT, Nn, 0);
    for (int r = 0; r < R; ++r) {
        zero_f32<<<(unsigned)(((long)Nn * DHID + 255) / 256), 256, 0, stream>>>(
            msum, (long)Nn * DHID);
        edge_aggregate<<<(unsigned)((aggT + 255) / 256), 256, 0, stream>>>(
            h, src, dst, et, E, r, msum);
        rgcn_gemm_strip<4><<<rowT, 32, 0, stream>>>(
            msum, DHID, bsw2 + (size_t)r * SW2, cnt, r, nullptr, outp, DOUT, Nn, 1);
    }
}
